// RotaryMultiheadAttention_32263794328291
// MI455X (gfx1250) — matrix operand order repaired and checked by reading, not compiled, not run on hardware
//
#include <hip/hip_runtime.h>

// ---------------- problem constants ----------------
static constexpr int Bc   = 2;
static constexpr int Lc   = 2048;
static constexpr int Ec   = 1024;
static constexpr int Hc   = 16;
static constexpr int HDc  = 64;      // head dim
static constexpr int HALFc= 32;      // rope half
static constexpr int N_QKV = 3 * Ec; // 3072

// ---------------- WMMA types ----------------
typedef __attribute__((ext_vector_type(8)))  float  v8f;
typedef __attribute__((ext_vector_type(16))) __bf16 v16bf;
typedef __attribute__((ext_vector_type(8)))  __bf16 v8bf;
typedef int v4i __attribute__((vector_size(16)));   // pointee type of async-LDS builtin

static __device__ __forceinline__ v16bf cat16(v8bf lo, v8bf hi) {
  return __builtin_shufflevector(lo, hi, 0,1,2,3,4,5,6,7,8,9,10,11,12,13,14,15);
}
static __device__ __forceinline__ v8bf ld8(const __bf16* p) {
  return *(const v8bf*)p;   // 16-byte aligned by construction
}
static __device__ __forceinline__ v8f vzero() {
  v8f z;
#pragma unroll
  for (int i = 0; i < 8; ++i) z[i] = 0.0f;
  return z;
}
#define WMMA_BF16(a, b, c) \
  __builtin_amdgcn_wmma_f32_16x16x32_bf16(false, (a), false, (b), (short)0, (c), false, false)

// ---------------- CDNA5 async global->LDS copy (ASYNCcnt) ----------------
#if defined(__has_builtin)
#if __has_builtin(__builtin_amdgcn_global_load_async_to_lds_b128) && \
    __has_builtin(__builtin_amdgcn_s_wait_asynccnt)
#define USE_ASYNC_LDS 1
#endif
#endif
#ifndef USE_ASYNC_LDS
#define USE_ASYNC_LDS 0
#endif

// copy 16 bytes global -> LDS for this lane
static __device__ __forceinline__ void cp16_g2l(const __bf16* g, __bf16* l) {
#if USE_ASYNC_LDS
  // builtin signature: (global int4*, local int4*, imm offset, imm cpol)
  __attribute__((address_space(1))) void* gp =
      (__attribute__((address_space(1))) void*)(void*)const_cast<__bf16*>(g);
  __attribute__((address_space(3))) void* lp =
      (__attribute__((address_space(3))) void*)(void*)l;
  __builtin_amdgcn_global_load_async_to_lds_b128(
      (__attribute__((address_space(1))) v4i*)gp,
      (__attribute__((address_space(3))) v4i*)lp, 0, 0);
#else
  *(uint4*)l = *(const uint4*)g;   // synchronous fallback
#endif
}
static __device__ __forceinline__ void wait_async(int n) {
#if USE_ASYNC_LDS
  if (n == 0) __builtin_amdgcn_s_wait_asynccnt(0);
  else        __builtin_amdgcn_s_wait_asynccnt(2);
#else
  (void)n;
#endif
}

// ---------------- stage 0: conversions ----------------
__global__ __launch_bounds__(256) void cvt_f32_bf16(const float* __restrict__ src,
                                                    __bf16* __restrict__ dst, int n) {
  int i = blockIdx.x * 256 + threadIdx.x;
  if (i < n) dst[i] = (__bf16)src[i];
}

// dst[n][K] = (bf16) src[k][N]
__global__ __launch_bounds__(256) void cvt_transpose_bf16(const float* __restrict__ src,
                                                          __bf16* __restrict__ dst,
                                                          int K, int N) {
  int idx = blockIdx.x * 256 + threadIdx.x;
  if (idx >= K * N) return;
  int k = idx / N, n = idx % N;
  dst[(size_t)n * K + k] = (__bf16)src[idx];
}

// ---------------- stage 1: QKV GEMM + RoPE epilogue ----------------
// grid: (N_QKV/256, (B*L)/32), block 256 (8 waves, 2(M) x 4(N)); wave tile 16x64
__global__ __launch_bounds__(256) void qkv_rope_kernel(
    const __bf16* __restrict__ xbf,    // [4096][1024]
    const __bf16* __restrict__ WqkvT,  // [3072][1024]
    const float*  __restrict__ bqkv,   // [3072]
    const float*  __restrict__ cosb,   // [L][32]
    const float*  __restrict__ sinb,   // [L][32]
    __bf16* __restrict__ Qb,           // [B*H][L][64]
    __bf16* __restrict__ Kb,           // [B*H][L][64]
    __bf16* __restrict__ Vt)           // [B*H][64][L]
{
  const int tid = threadIdx.x;
  const int l   = tid & 31;
  const int w   = tid >> 5;
  const int lm  = l & 15;
  const int lh  = l >> 4;

  const int rowbase = blockIdx.y * 32 + (w >> 2) * 16;
  const int nbase   = blockIdx.x * 256 + (w & 3) * 64;

  v8f acc[4];
#pragma unroll
  for (int j = 0; j < 4; ++j) acc[j] = vzero();

  const __bf16* arow = xbf + (size_t)(rowbase + lm) * Ec;
  for (int k0 = 0; k0 < Ec; k0 += 32) {
    v16bf a = cat16(ld8(arow + k0 + 8 * lh), ld8(arow + k0 + 16 + 8 * lh));
#pragma unroll
    for (int j = 0; j < 4; ++j) {
      const __bf16* bp = WqkvT + (size_t)(nbase + 16 * j + lm) * Ec + k0 + 8 * lh;
      v16bf b = cat16(ld8(bp), ld8(bp + 16));
      acc[j] = WMMA_BF16(a, b, acc[j]);
    }
  }

  const int sec  = nbase / Ec;            // 0=q 1=k 2=v
  const int hh   = (nbase % Ec) / HDc;    // head
  const int brow = rowbase / Lc;
  const int posbase = rowbase % Lc;

  float bias[4];
#pragma unroll
  for (int j = 0; j < 4; ++j) bias[j] = bqkv[nbase + 16 * j + lm];

  if (sec < 2) {
    __bf16* dst = (sec == 0 ? Qb : Kb) + ((size_t)(brow * Hc + hh) * Lc) * HDc;
#pragma unroll
    for (int jv = 0; jv < 8; ++jv) {
      const int pos = posbase + jv + 8 * lh;
#pragma unroll
      for (int j = 0; j < 2; ++j) {
        const int f = 16 * j + lm;
        const float c = cosb[pos * HALFc + f];
        const float s = sinb[pos * HALFc + f];
        const float t1 = acc[j][jv]     + bias[j];
        const float t2 = acc[j + 2][jv] + bias[j + 2];
        dst[(size_t)pos * HDc + f]          = (__bf16)(t1 * c - t2 * s);
        dst[(size_t)pos * HDc + HALFc + f]  = (__bf16)(t1 * s + t2 * c);
      }
    }
  } else {
    __bf16* dst = Vt + ((size_t)(brow * Hc + hh) * HDc) * Lc;  // [d][pos]
#pragma unroll
    for (int jv = 0; jv < 8; ++jv) {
      const int pos = posbase + jv + 8 * lh;
#pragma unroll
      for (int j = 0; j < 4; ++j) {
        const int d = 16 * j + lm;
        dst[(size_t)d * Lc + pos] = (__bf16)(acc[j][jv] + bias[j]);
      }
    }
  }
}

// ---------------- stage 2: flash attention ----------------
// grid: (B*H, L/128), block 256 (8 waves); wave = 16 query rows; 32 keys / iter.
// K/V tiles staged in LDS once per block (8x traffic cut) with async double-buffer.
__global__ __launch_bounds__(256) void flash_attn_kernel(
    const __bf16* __restrict__ Qb,  // [B*H][L][64]
    const __bf16* __restrict__ Kb,  // [B*H][L][64]
    const __bf16* __restrict__ Vt,  // [B*H][64][L]
    __bf16* __restrict__ Ob)        // [B*L][1024]
{
  __shared__ __align__(16) __bf16 Kt[2][32 * 64];    // [key][d]   4 KB x2
  __shared__ __align__(16) __bf16 Vtl[2][64 * 32];   // [d][key]   4 KB x2
  __shared__ __align__(16) __bf16 Pst[8][16 * 32];   // per-wave P tile

  const int tid = threadIdx.x;
  const int l   = tid & 31;
  const int w   = tid >> 5;
  const int lm  = l & 15;
  const int lh  = l >> 4;

  const int bh    = blockIdx.x;
  const int qbase = blockIdx.y * 128 + w * 16;

  const __bf16* Qp = Qb + (size_t)bh * Lc * HDc;
  const __bf16* Kp = Kb + (size_t)bh * Lc * HDc;
  const __bf16* Vp = Vt + (size_t)bh * HDc * Lc;

  // per-thread copy slots: K tile = contiguous 4KB; V tile = 64 rows x 64B
  const int kcpy = tid * 8;            // element offset in 2048-elem K tile
  const int vrow = tid >> 2;           // d row 0..63
  const int vcol = (tid & 3) * 8;      // 16B chunk within 32-key row

  // Q fragments (A-layout)
  const __bf16* qrow = Qp + (size_t)(qbase + lm) * HDc;
  const v16bf qa0 = cat16(ld8(qrow + 8 * lh),      ld8(qrow + 16 + 8 * lh));
  const v16bf qa1 = cat16(ld8(qrow + 32 + 8 * lh), ld8(qrow + 48 + 8 * lh));

  v8f o[4];
#pragma unroll
  for (int j = 0; j < 4; ++j) o[j] = vzero();
  float rmax[8], rsum[8];
#pragma unroll
  for (int jv = 0; jv < 8; ++jv) { rmax[jv] = -INFINITY; rsum[jv] = 0.0f; }

  const float scale = 0.125f;
  const int   ntiles = Lc / 32;   // 64

  // prologue: stage tile 0 into buffer 0
  cp16_g2l(Kp + kcpy, &Kt[0][kcpy]);
  cp16_g2l(Vp + (size_t)vrow * Lc + vcol, &Vtl[0][vrow * 32 + vcol]);

  for (int i = 0; i < ntiles; ++i) {
    const int cur = i & 1;
    if (i + 1 < ntiles) {                     // prefetch next tile
      const int nb = (i + 1) & 1;
      const int kt = (i + 1) * 32;
      cp16_g2l(Kp + (size_t)kt * HDc + kcpy, &Kt[nb][kcpy]);
      cp16_g2l(Vp + (size_t)vrow * Lc + kt + vcol, &Vtl[nb][vrow * 32 + vcol]);
      wait_async(2);                          // drain tile i, keep i+1 in flight
    } else {
      wait_async(0);
    }
    __syncthreads();                          // tile i visible to all waves

    // ---- S = Q @ K^T for 32 keys from LDS ----
    const __bf16* kb = &Kt[cur][0];
    const __bf16* kr0 = kb + lm * 64 + 8 * lh;           // keys 0..15
    const __bf16* kr1 = kb + (16 + lm) * 64 + 8 * lh;    // keys 16..31
    v16bf bk0lo = cat16(ld8(kr0),      ld8(kr0 + 16));
    v16bf bk0hi = cat16(ld8(kr0 + 32), ld8(kr0 + 48));
    v16bf bk1lo = cat16(ld8(kr1),      ld8(kr1 + 16));
    v16bf bk1hi = cat16(ld8(kr1 + 32), ld8(kr1 + 48));

    v8f s0 = vzero(), s1 = vzero();
    s0 = WMMA_BF16(qa0, bk0lo, s0);
    s0 = WMMA_BF16(qa1, bk0hi, s0);
    s1 = WMMA_BF16(qa0, bk1lo, s1);
    s1 = WMMA_BF16(qa1, bk1hi, s1);

    // ---- online softmax (rows live in 16-lane half-groups) ----
#pragma unroll
    for (int jv = 0; jv < 8; ++jv) {
      const float a  = s0[jv] * scale;
      const float b2 = s1[jv] * scale;
      float t = fmaxf(a, b2);
#pragma unroll
      for (int off = 1; off < 16; off <<= 1) t = fmaxf(t, __shfl_xor(t, off, 16));
      const float nm   = fmaxf(rmax[jv], t);
      const float corr = __expf(rmax[jv] - nm);
      rmax[jv] = nm;
      const float p0 = __expf(a - nm);
      const float p1 = __expf(b2 - nm);
      float ps = p0 + p1;
#pragma unroll
      for (int off = 1; off < 16; off <<= 1) ps += __shfl_xor(ps, off, 16);
      rsum[jv] = rsum[jv] * corr + ps;
#pragma unroll
      for (int j = 0; j < 4; ++j) o[j][jv] *= corr;
      Pst[w][(jv + 8 * lh) * 32 + lm]      = (__bf16)p0;
      Pst[w][(jv + 8 * lh) * 32 + 16 + lm] = (__bf16)p1;
    }
    __syncthreads();

    // ---- P A-fragment from LDS, O += P @ V (V from LDS) ----
    const __bf16* pr = &Pst[w][lm * 32 + 8 * lh];
    v16bf pa = cat16(*(const v8bf*)pr, *(const v8bf*)(pr + 16));

    const __bf16* vb = &Vtl[cur][0];
#pragma unroll
    for (int j = 0; j < 4; ++j) {
      const __bf16* vr = vb + (16 * j + lm) * 32 + 8 * lh;
      v16bf vfrag = cat16(ld8(vr), ld8(vr + 16));
      o[j] = WMMA_BF16(pa, vfrag, o[j]);
    }
    __syncthreads();   // protects buffer 'cur' before it is re-staged at i+2
  }

  // ---- finalize & store ----
  const int hh   = bh % Hc;
  const int brow = bh / Hc;
#pragma unroll
  for (int jv = 0; jv < 8; ++jv) {
    const float inv = 1.0f / rsum[jv];
    const int row   = qbase + jv + 8 * lh;
#pragma unroll
    for (int j = 0; j < 4; ++j) {
      Ob[(size_t)(brow * Lc + row) * Ec + hh * HDc + 16 * j + lm] =
          (__bf16)(o[j][jv] * inv);
    }
  }
}

// ---------------- stage 3: output projection ----------------
__global__ __launch_bounds__(256) void outproj_kernel(
    const __bf16* __restrict__ Ob,     // [4096][1024]
    const __bf16* __restrict__ WoutT,  // [1024][1024]
    const float*  __restrict__ bout,   // [1024]
    float* __restrict__ out)           // [4096][1024]
{
  const int tid = threadIdx.x;
  const int l   = tid & 31;
  const int w   = tid >> 5;
  const int lm  = l & 15;
  const int lh  = l >> 4;

  const int rowbase = blockIdx.y * 32 + (w >> 2) * 16;
  const int nbase   = blockIdx.x * 256 + (w & 3) * 64;

  v8f acc[4];
#pragma unroll
  for (int j = 0; j < 4; ++j) acc[j] = vzero();

  const __bf16* arow = Ob + (size_t)(rowbase + lm) * Ec;
  for (int k0 = 0; k0 < Ec; k0 += 32) {
    v16bf a = cat16(ld8(arow + k0 + 8 * lh), ld8(arow + k0 + 16 + 8 * lh));
#pragma unroll
    for (int j = 0; j < 4; ++j) {
      const __bf16* bp = WoutT + (size_t)(nbase + 16 * j + lm) * Ec + k0 + 8 * lh;
      v16bf b = cat16(ld8(bp), ld8(bp + 16));
      acc[j] = WMMA_BF16(a, b, acc[j]);
    }
  }

#pragma unroll
  for (int jv = 0; jv < 8; ++jv) {
    const int row = rowbase + jv + 8 * lh;
#pragma unroll
    for (int j = 0; j < 4; ++j) {
      const int n = nbase + 16 * j + lm;
      out[(size_t)row * Ec + n] = acc[j][jv] + bout[n];
    }
  }
}

// ---------------- launcher ----------------
extern "C" void kernel_launch(void* const* d_in, const int* in_sizes, int n_in,
                              void* d_out, int out_size, void* d_ws, size_t ws_size,
                              hipStream_t stream) {
  const float* x    = (const float*)d_in[0];
  const float* cosb = (const float*)d_in[1];
  const float* sinb = (const float*)d_in[2];
  const float* Wqkv = (const float*)d_in[3];
  const float* bqkv = (const float*)d_in[4];
  const float* Wout = (const float*)d_in[5];
  const float* bout = (const float*)d_in[6];
  float* out = (float*)d_out;

  const int M = Bc * Lc;  // 4096

  char* ws = (char*)d_ws;
  __bf16* xbf   = (__bf16*)ws;  ws += (size_t)M * Ec * 2;
  __bf16* WqkvT = (__bf16*)ws;  ws += (size_t)N_QKV * Ec * 2;
  __bf16* WoutT = (__bf16*)ws;  ws += (size_t)Ec * Ec * 2;
  __bf16* Qb    = (__bf16*)ws;  ws += (size_t)Bc * Hc * Lc * HDc * 2;
  __bf16* Kb    = (__bf16*)ws;  ws += (size_t)Bc * Hc * Lc * HDc * 2;
  __bf16* Vt    = (__bf16*)ws;  ws += (size_t)Bc * Hc * Lc * HDc * 2;
  __bf16* Ob    = (__bf16*)ws;  ws += (size_t)M * Ec * 2;
  (void)ws_size; (void)in_sizes; (void)n_in; (void)out_size;

  {
    int n = M * Ec;
    cvt_f32_bf16<<<(n + 255) / 256, 256, 0, stream>>>(x, xbf, n);
  }
  cvt_transpose_bf16<<<(Ec * N_QKV + 255) / 256, 256, 0, stream>>>(Wqkv, WqkvT, Ec, N_QKV);
  cvt_transpose_bf16<<<(Ec * Ec + 255) / 256, 256, 0, stream>>>(Wout, WoutT, Ec, Ec);

  {
    dim3 grid(N_QKV / 256, M / 32);
    qkv_rope_kernel<<<grid, 256, 0, stream>>>(xbf, WqkvT, bqkv, cosb, sinb, Qb, Kb, Vt);
  }
  {
    dim3 grid(Bc * Hc, Lc / 128);
    flash_attn_kernel<<<grid, 256, 0, stream>>>(Qb, Kb, Vt, Ob);
  }
  {
    dim3 grid(Ec / 256, M / 32);
    outproj_kernel<<<grid, 256, 0, stream>>>(Ob, WoutT, bout, out);
  }
}
